// Model_40956808134827
// MI455X (gfx1250) — matrix operand order repaired and checked by reading, not compiled, not run on hardware
//
#include <hip/hip_runtime.h>
#include <hip/hip_bf16.h>
#include <math.h>

// ---------------------------------------------------------------------------
// MI455X (gfx1250) implementation.
// All heavy GEMMs (conv-as-GEMM + FC) use v_wmma_f32_16x16x32_bf16 with fp32
// accumulation; each wave computes a 16x64 output tile (A-fragment reuse x4).
// Tail ops (attention, sigmoid scores, |x-mu|^p, top-k, gathers) are fp32 VALU.
// ---------------------------------------------------------------------------

typedef __attribute__((ext_vector_type(16))) __bf16 v16bf;
typedef __attribute__((ext_vector_type(8)))  float  v8f;
typedef __hip_bfloat16 bf16_t;

union FragU { uint4 q[2]; v16bf v; };

__device__ __forceinline__ bf16_t to_bf(float f) { return __float2bfloat16(f); }

// ------------------------------ prep kernels -------------------------------

// x (640,32,2048) fp32 -> xpad (640,40,2048) bf16, rows 0..3 / 36..39 zero.
__global__ void prep_xpad_kernel(const float* __restrict__ x, bf16_t* __restrict__ xpad) {
    size_t idx = (size_t)blockIdx.x * 256 + threadIdx.x;
    int f = (int)(idx & 2047);
    size_t rem = idx >> 11;          // b*40 + row
    int row = (int)(rem % 40);
    size_t b = rem / 40;
    int t = row - 4;
    float v = (t >= 0 && t < 32) ? x[((b * 32 + (size_t)t) << 11) + f] : 0.0f;
    xpad[idx] = to_bf(v);
}

// zero hpad rows 0 and 33 for all 640 samples
__global__ void hzero_kernel(bf16_t* __restrict__ hpad) {
    size_t idx = (size_t)blockIdx.x * 256 + threadIdx.x;   // 640*2*2048
    int f = (int)(idx & 2047);
    size_t rem = idx >> 11;
    int r2 = (int)(rem & 1);
    size_t b = rem >> 1;
    int row = r2 ? 33 : 0;
    hpad[((b * 34 + (size_t)row) << 11) + f] = to_bf(0.0f);
}

// weights (O,I,3) fp32 -> (O,3,I) bf16
__global__ void w3_kernel(const float* __restrict__ w, bf16_t* __restrict__ o, int O, int I) {
    size_t idx = (size_t)blockIdx.x * 256 + threadIdx.x;
    if (idx >= (size_t)O * 3 * I) return;
    int i  = (int)(idx % I);
    int k  = (int)((idx / I) % 3);
    int oo = (int)(idx / ((size_t)3 * I));
    o[idx] = to_bf(w[((size_t)oo * I + i) * 3 + k]);
}

// fp32 -> bf16 copy
__global__ void w2_kernel(const float* __restrict__ w, bf16_t* __restrict__ o, size_t n) {
    size_t idx = (size_t)blockIdx.x * 256 + threadIdx.x;
    if (idx >= n) return;
    o[idx] = to_bf(w[idx]);
}

// ------------------------------ WMMA GEMM ----------------------------------
// Out[M,N] = A[M,K](bf16) x Bt[N,K](bf16)^T, epilogue options.
// CONV=1: A is padded activation (B,TP,2048) bf16; K = taps*2048,
//         row for (m,tap) = b*TP + t + pad_off + (tap-1)*dil.
// relu_mode: 0 none; 1 relu; 2 relu(relu(acc)+xres)   (c5 -> feat)

template<int CONV>
__global__ __launch_bounds__(256) void gemm_wmma_kernel(
    const bf16_t* __restrict__ A, const bf16_t* __restrict__ Bt,
    float* __restrict__ outF, bf16_t* __restrict__ outB,
    const float* __restrict__ bias, const float* __restrict__ resid,
    const float* __restrict__ xres,
    int M, int N, int K,
    int dil, int TP, int pad_off,
    int out_stride, int out_coloff, int out_tp, int relu_mode)
{
    const int wave = threadIdx.x >> 5;
    const int lane = threadIdx.x & 31;
    const int half = lane >> 4;
    const int l15  = lane & 15;
    const int m0 = (blockIdx.y * 8 + wave) * 16;
    const int n0 = blockIdx.x * 64;
    if (m0 >= M) return;                  // wave-uniform, EXEC stays full
    const int am = m0 + l15;              // A row owned by this lane

    v8f acc0 = {0,0,0,0,0,0,0,0};
    v8f acc1 = acc0, acc2 = acc0, acc3 = acc0;

    const bf16_t* bbase = Bt + (size_t)(n0 + l15) * K;

    for (int k0 = 0; k0 < K; k0 += 32) {
        // ---- A fragment: 16 contiguous K per lane at k0 + half*16
        FragU a;
        const bf16_t* ap;
        if (CONV) {
            const int tap = k0 >> 11;        // K chunk of 32 never straddles a tap
            const int f0  = k0 & 2047;
            const int b   = am >> 5;
            const int t   = am & 31;
            ap = A + (((size_t)(b * TP + t + pad_off + (tap - 1) * dil)) << 11) + f0 + half * 16;
        } else {
            ap = A + (size_t)am * K + k0 + half * 16;
        }
        a.q[0] = *(const uint4*)(ap);
        a.q[1] = *(const uint4*)(ap + 8);

        // ---- B fragments: 16 contiguous K per lane at k0 + half*16
        const bf16_t* bp = bbase + k0 + half * 16;
        FragU b0, b1, b2, b3;
        b0.q[0] = *(const uint4*)(bp);
        b0.q[1] = *(const uint4*)(bp + 8);
        b1.q[0] = *(const uint4*)(bp + (size_t)16 * K);
        b1.q[1] = *(const uint4*)(bp + (size_t)16 * K + 8);
        b2.q[0] = *(const uint4*)(bp + (size_t)32 * K);
        b2.q[1] = *(const uint4*)(bp + (size_t)32 * K + 8);
        b3.q[0] = *(const uint4*)(bp + (size_t)48 * K);
        b3.q[1] = *(const uint4*)(bp + (size_t)48 * K + 8);

        acc0 = __builtin_amdgcn_wmma_f32_16x16x32_bf16(false, a.v, false, b0.v, (short)0, acc0, false, false);
        acc1 = __builtin_amdgcn_wmma_f32_16x16x32_bf16(false, a.v, false, b1.v, (short)0, acc1, false, false);
        acc2 = __builtin_amdgcn_wmma_f32_16x16x32_bf16(false, a.v, false, b2.v, (short)0, acc2, false, false);
        acc3 = __builtin_amdgcn_wmma_f32_16x16x32_bf16(false, a.v, false, b3.v, (short)0, acc3, false, false);
    }

    v8f accs[4] = {acc0, acc1, acc2, acc3};
#pragma unroll
    for (int j = 0; j < 4; ++j) {
        const int col = n0 + j * 16 + l15;
        const float bv = bias ? bias[col] : 0.0f;
#pragma unroll
        for (int r = 0; r < 8; ++r) {
            const int m = m0 + half * 8 + r;     // C/D layout: m = half*8 + vgpr
            float v = accs[j][r] + bv;
            if (resid) v += resid[(size_t)m * N + col];
            if (relu_mode == 1)      v = fmaxf(v, 0.0f);
            else if (relu_mode == 2) { v = fmaxf(v, 0.0f) + xres[(size_t)m * N + col]; v = fmaxf(v, 0.0f); }
            const size_t orow = out_tp ? ((size_t)(m >> 5) * out_tp + (m & 31) + 1) : (size_t)m;
            const size_t oidx = orow * out_stride + out_coloff + col;
            if (outF) outF[oidx] = v;
            if (outB) outB[oidx] = to_bf(v);
        }
    }
}

// --------------------------- attention (fp32) ------------------------------
// tpg (20480,768) fp32: [0:256)=theta rows, [256:512)=phi, [512:768)=g.
// att[t,s] = dot(th_t, ph_s)/32 ;  y[t,c] = sum_s att[t,s]*g[s,c] -> ybf bf16
__global__ void attn_kernel(const float* __restrict__ tpg, bf16_t* __restrict__ ybf) {
    const int b = blockIdx.x;                       // 640
    __shared__ float att[32][32];
    const float* base = tpg + (size_t)b * 32 * 768;
    for (int p = threadIdx.x; p < 1024; p += 256) {
        int t = p >> 5, s = p & 31;
        const float* th = base + (size_t)t * 768;
        const float* ph = base + (size_t)s * 768 + 256;
        float acc = 0.0f;
        for (int c = 0; c < 256; ++c) acc += th[c] * ph[c];
        att[t][s] = acc * (1.0f / 32.0f);
    }
    __syncthreads();
    for (int p = threadIdx.x; p < 32 * 256; p += 256) {
        int t = p >> 8, c = p & 255;
        float acc = 0.0f;
        for (int s = 0; s < 32; ++s) acc += att[t][s] * base[(size_t)s * 768 + 512 + c];
        ybf[((size_t)b * 32 + t) * 256 + c] = to_bf(acc);
    }
}

// ------------------------------ scores -------------------------------------
// s3 = sigmoid(s2 @ fc3 + b); scores[bb,t] = mean over 10 crops.
__global__ void scores_kernel(const float* __restrict__ s2, const float* __restrict__ fc3w,
                              const float* __restrict__ fc3b, float* __restrict__ scores,
                              float* __restrict__ outAll) {
    const int bb = blockIdx.x;   // 64
    const int t  = threadIdx.x;  // 32
    float m = 0.0f;
    for (int c = 0; c < 10; ++c) {
        const float* row = s2 + ((size_t)(bb * 10 + c) * 32 + t) * 128;
        float z = fc3b[0];
        for (int i = 0; i < 128; ++i) z += row[i] * fc3w[i];
        m += 1.0f / (1.0f + expf(-z));
    }
    m *= 0.1f;
    scores[bb * 32 + t] = m;
    outAll[bb * 32 + t] = m;
}

// mu[b,f] = mean_t feat[b,t,f]
__global__ void mu_kernel(const float* __restrict__ feat, float* __restrict__ mu) {
    const int f = blockIdx.x * 256 + threadIdx.x;
    const int b = blockIdx.y;     // 640
    float s = 0.0f;
    for (int t = 0; t < 32; ++t) s += feat[(((size_t)b * 32 + t) << 11) + f];
    mu[((size_t)b << 11) + f] = s * (1.0f / 32.0f);
}

// af2[nn,t,f] = mean_c feat[320+nn*10+c, t, f]
__global__ void af2_kernel(const float* __restrict__ feat, float* __restrict__ af2) {
    const int f = blockIdx.x * 256 + threadIdx.x;
    const int nn = blockIdx.y >> 5;
    const int t  = blockIdx.y & 31;
    float s = 0.0f;
    for (int c = 0; c < 10; ++c)
        s += feat[(((size_t)(320 + nn * 10 + c) * 32 + t) << 11) + f];
    af2[(((size_t)nn * 32 + t) << 11) + f] = s * 0.1f;
}

// diff_seg on af2, top-3 idx per nn (shared by abn/nor), and score outputs.
__global__ void diff_kernel(const float* __restrict__ af2, const float* __restrict__ scores,
                            int* __restrict__ idxbuf, float* __restrict__ outSAbn,
                            float* __restrict__ outSNor) {
    const int nn = blockIdx.x;            // 32
    const int s  = threadIdx.x;           // 32
    const int t  = threadIdx.y;           // 32
    __shared__ float dotm[32][32];
    __shared__ float diff[32];
    const float* base = af2 + ((size_t)nn * 32) * 2048;
    float acc = 0.0f;
    const float* rt = base + (size_t)t * 2048;
    const float* rs = base + (size_t)s * 2048;
    for (int f = 0; f < 2048; ++f) acc += rt[f] * rs[f];
    dotm[t][s] = acc;
    __syncthreads();
    if (s == 0) {
        float b0 = 1e30f, b1 = 1e30f, b2 = 1e30f;   // 3 smallest D
        const float x2t = dotm[t][t];
        for (int ss = 0; ss < 32; ++ss) {
            float D = x2t + dotm[ss][ss] - 2.0f * dotm[t][ss];
            if (ss == t) D += 1000000000.0f;
            if (D < b2) {
                if (D < b1) { b2 = b1; if (D < b0) { b1 = b0; b0 = D; } else b1 = D; }
                else b2 = D;
            }
        }
        diff[t] = (b0 + b1 + b2) * (1.0f / 3.0f);
    }
    __syncthreads();
    if (s == 0 && t == 0) {
        bool used[32];
        for (int i = 0; i < 32; ++i) used[i] = false;
        float sa = 0.0f, sn = 0.0f;
        for (int j = 0; j < 3; ++j) {
            float bv = -1e30f; int bi = 0;
            for (int i = 0; i < 32; ++i)
                if (!used[i] && diff[i] > bv) { bv = diff[i]; bi = i; }  // ties -> lowest idx
            used[bi] = true;
            idxbuf[nn * 3 + j] = bi;
            sa += scores[(32 + nn) * 32 + bi];
            sn += scores[nn * 32 + bi];
        }
        outSAbn[nn] = sa * (1.0f / 3.0f);
        outSNor[nn] = sn * (1.0f / 3.0f);
    }
}

__device__ __forceinline__ void insert4(float& a, float& b, float& c, float& d, float v) {
    if (v > d) {
        if (v > b) {
            if (v > a) { d = c; c = b; b = a; a = v; }
            else       { d = c; c = b; b = v; }
        } else {
            if (v > c) { d = c; c = v; }
            else       d = v;
        }
    }
}

// per-(row,t) top-4 of |feat-mu|^p over F=2048 (desc). grp0: rows 320+, p=6.
__global__ void topk4_kernel(const float* __restrict__ feat, const float* __restrict__ mu,
                             float* __restrict__ out) {
    const int grp  = blockIdx.y;           // 0 abn(p6), 1 nor(p5)
    const int i    = blockIdx.x;           // 0..319
    const int wave = threadIdx.x >> 5;
    const int lane = threadIdx.x & 31;
    const int brow = (grp == 0 ? 320 : 0) + i;
    const float* mrow = mu + ((size_t)brow << 11);
    for (int ti = 0; ti < 4; ++ti) {
        const int t = wave * 4 + ti;
        const float* frow = feat + (((size_t)brow * 32 + t) << 11);
        float v0 = -1.0f, v1 = -1.0f, v2 = -1.0f, v3 = -1.0f;
        for (int f = lane; f < 2048; f += 32) {
            float d  = fabsf(frow[f] - mrow[f]);
            float d2 = d * d;
            float val = (grp == 0) ? d2 * d2 * d2 : d2 * d2 * d;
            insert4(v0, v1, v2, v3, val);
        }
        for (int off = 16; off > 0; off >>= 1) {
            float w0 = __shfl_down(v0, off), w1 = __shfl_down(v1, off);
            float w2 = __shfl_down(v2, off), w3 = __shfl_down(v3, off);
            insert4(v0, v1, v2, v3, w0);
            insert4(v0, v1, v2, v3, w1);
            insert4(v0, v1, v2, v3, w2);
            insert4(v0, v1, v2, v3, w3);
        }
        if (lane == 0) {
            size_t o = (grp == 0 ? (size_t)64 : (size_t)41024) + ((size_t)i * 32 + t) * 4;
            out[o] = v0; out[o + 1] = v1; out[o + 2] = v2; out[o + 3] = v3;
        }
    }
}

// feat_abn / feat_nor gathers: |feat[row, idx[nn,j], f] - mu[row, f]|^p
__global__ void gather_kernel(const float* __restrict__ feat, const float* __restrict__ mu,
                              const int* __restrict__ idx, float* __restrict__ out) {
    const int grp = blockIdx.y;            // 0 abn(p6), 1 nor(p5)
    const int p   = blockIdx.x;            // 0..959 = r*3+j, r = c*32+nn
    const int j   = p % 3;
    const int r   = p / 3;
    const int c   = r >> 5;
    const int nn  = r & 31;
    const int tt  = idx[nn * 3 + j];
    const int brow = (grp == 0 ? 320 : 0) + nn * 10 + c;
    const float* frow = feat + (((size_t)brow * 32 + tt) << 11);
    const float* mrow = mu + ((size_t)brow << 11);
    const size_t base = (grp == 0 ? (size_t)81984 : (size_t)2048064) + (size_t)p * 2048;
    for (int f = threadIdx.x; f < 2048; f += 256) {
        float d  = fabsf(frow[f] - mrow[f]);
        float d2 = d * d;
        out[base + f] = (grp == 0) ? d2 * d2 * d2 : d2 * d2 * d;
    }
}

// ------------------------------ launch -------------------------------------

// workspace byte offsets (all 256B aligned). FEATF aliases dead XPAD+TPGO.
static const size_t OF_XPAD  = 0;                         // 104,857,600 B
static const size_t OF_TPGO  = OF_XPAD + 104857600;       //  62,914,560 B
static const size_t OF_FEATF = 0;                         // 167,772,160 B (alias)
static const size_t OF_HPAD  = 167772160;                 //  89,128,960 B
static const size_t OF_C1B   = OF_HPAD + 89128960;
static const size_t OF_C2B   = OF_C1B + 6291456;
static const size_t OF_C3B   = OF_C2B + 6291456;
static const size_t OF_C5B   = OF_C3B + 6291456;
static const size_t OF_C4B   = OF_C5B + 25165824;
static const size_t OF_TPGW  = OF_C4B + 2097152;
static const size_t OF_WB    = OF_TPGW + 786432;
static const size_t OF_FC1W  = OF_WB + 262144;
static const size_t OF_FC2W  = OF_FC1W + 2097152;
static const size_t OF_O4F   = OF_FC2W + 131072;
static const size_t OF_O4B   = OF_O4F + 41943040;
static const size_t OF_YBF   = OF_O4B + 20971520;
static const size_t OF_FEATB = OF_YBF + 10485760;
static const size_t OF_S1B   = OF_FEATB + 83886080;
static const size_t OF_S2F   = OF_S1B + 20971520;
static const size_t OF_SCORE = OF_S2F + 10485760;
static const size_t OF_MU    = OF_SCORE + 8192;
static const size_t OF_AF2   = OF_MU + 5242880;
static const size_t OF_IDX   = OF_AF2 + 8388608;          // end ~ 508.7 MB

extern "C" void kernel_launch(void* const* d_in, const int* in_sizes, int n_in,
                              void* d_out, int out_size, void* d_ws, size_t ws_size,
                              hipStream_t stream) {
    (void)in_sizes; (void)n_in; (void)out_size; (void)ws_size;
    const float* x     = (const float*)d_in[0];
    const float* c1    = (const float*)d_in[1];
    const float* c2    = (const float*)d_in[2];
    const float* c3    = (const float*)d_in[3];
    const float* c4    = (const float*)d_in[4];
    const float* theta = (const float*)d_in[5];
    const float* phi   = (const float*)d_in[6];
    const float* g     = (const float*)d_in[7];
    const float* Wnl   = (const float*)d_in[8];
    const float* c5    = (const float*)d_in[9];
    const float* fc1w  = (const float*)d_in[10];
    const float* fc1b  = (const float*)d_in[11];
    const float* fc2w  = (const float*)d_in[12];
    const float* fc2b  = (const float*)d_in[13];
    const float* fc3w  = (const float*)d_in[14];
    const float* fc3b  = (const float*)d_in[15];
    float* out = (float*)d_out;

    char* ws = (char*)d_ws;
    bf16_t* xpad  = (bf16_t*)(ws + OF_XPAD);
    float*  tpgo  = (float*) (ws + OF_TPGO);
    float*  featf = (float*) (ws + OF_FEATF);
    bf16_t* hpad  = (bf16_t*)(ws + OF_HPAD);
    bf16_t* c1b   = (bf16_t*)(ws + OF_C1B);
    bf16_t* c2b   = (bf16_t*)(ws + OF_C2B);
    bf16_t* c3b   = (bf16_t*)(ws + OF_C3B);
    bf16_t* c5b   = (bf16_t*)(ws + OF_C5B);
    bf16_t* c4b   = (bf16_t*)(ws + OF_C4B);
    bf16_t* tpgw  = (bf16_t*)(ws + OF_TPGW);
    bf16_t* wb    = (bf16_t*)(ws + OF_WB);
    bf16_t* wfc1  = (bf16_t*)(ws + OF_FC1W);
    bf16_t* wfc2  = (bf16_t*)(ws + OF_FC2W);
    float*  o4f   = (float*) (ws + OF_O4F);
    bf16_t* o4b   = (bf16_t*)(ws + OF_O4B);
    bf16_t* ybf   = (bf16_t*)(ws + OF_YBF);
    bf16_t* featb = (bf16_t*)(ws + OF_FEATB);
    bf16_t* s1b   = (bf16_t*)(ws + OF_S1B);
    float*  s2f   = (float*) (ws + OF_S2F);
    float*  scor  = (float*) (ws + OF_SCORE);
    float*  mu    = (float*) (ws + OF_MU);
    float*  af2   = (float*) (ws + OF_AF2);
    int*    idxb  = (int*)   (ws + OF_IDX);

    // ---- prep ----
    prep_xpad_kernel<<<204800, 256, 0, stream>>>(x, xpad);
    hzero_kernel<<<10240, 256, 0, stream>>>(hpad);
    w3_kernel<<<12288, 256, 0, stream>>>(c1, c1b, 512, 2048);
    w3_kernel<<<12288, 256, 0, stream>>>(c2, c2b, 512, 2048);
    w3_kernel<<<12288, 256, 0, stream>>>(c3, c3b, 512, 2048);
    w3_kernel<<<49152, 256, 0, stream>>>(c5, c5b, 2048, 2048);
    w2_kernel<<<4096, 256, 0, stream>>>(c4, c4b, (size_t)1048576);
    w2_kernel<<<512, 256, 0, stream>>>(theta, tpgw,          (size_t)131072);
    w2_kernel<<<512, 256, 0, stream>>>(phi,   tpgw + 131072, (size_t)131072);
    w2_kernel<<<512, 256, 0, stream>>>(g,     tpgw + 262144, (size_t)131072);
    w2_kernel<<<512, 256, 0, stream>>>(Wnl, wb, (size_t)131072);
    w2_kernel<<<4096, 256, 0, stream>>>(fc1w, wfc1, (size_t)1048576);
    w2_kernel<<<256, 256, 0, stream>>>(fc2w, wfc2, (size_t)65536);

    const int M = 20480;
    dim3 blk(256);
    // o1,o2,o3 -> hpad columns 0/512/1024 (relu, bf16)
    gemm_wmma_kernel<1><<<dim3(8, 160), blk, 0, stream>>>(xpad, c1b, nullptr, hpad,
        nullptr, nullptr, nullptr, M, 512, 6144, 1, 40, 4, 2048, 0,    34, 1);
    gemm_wmma_kernel<1><<<dim3(8, 160), blk, 0, stream>>>(xpad, c2b, nullptr, hpad,
        nullptr, nullptr, nullptr, M, 512, 6144, 2, 40, 4, 2048, 512,  34, 1);
    gemm_wmma_kernel<1><<<dim3(8, 160), blk, 0, stream>>>(xpad, c3b, nullptr, hpad,
        nullptr, nullptr, nullptr, M, 512, 6144, 4, 40, 4, 2048, 1024, 34, 1);
    // o4 (k=1 conv via dil=0), relu, fp32 + bf16
    gemm_wmma_kernel<1><<<dim3(8, 160), blk, 0, stream>>>(xpad, c4b, o4f, o4b,
        nullptr, nullptr, nullptr, M, 512, 2048, 0, 40, 4, 512, 0, 0, 1);
    // theta|phi|g fused (N=768), fp32
    gemm_wmma_kernel<0><<<dim3(12, 160), blk, 0, stream>>>(o4b, tpgw, tpgo, nullptr,
        nullptr, nullptr, nullptr, M, 768, 512, 0, 0, 0, 768, 0, 0, 0);
    // attention -> ybf
    attn_kernel<<<640, 256, 0, stream>>>(tpgo, ybf);
    // nl = y@W^T + o4 -> hpad columns 1536 (bf16)
    gemm_wmma_kernel<0><<<dim3(8, 160), blk, 0, stream>>>(ybf, wb, nullptr, hpad,
        nullptr, o4f, nullptr, M, 512, 256, 0, 0, 0, 2048, 1536, 34, 0);
    // c5 conv: feat = relu(relu(conv)+x) -> featf + featb
    gemm_wmma_kernel<1><<<dim3(32, 160), blk, 0, stream>>>(hpad, c5b, featf, featb,
        nullptr, nullptr, x, M, 2048, 6144, 1, 34, 1, 2048, 0, 0, 2);
    // fc1 (bias+relu) -> s1b
    gemm_wmma_kernel<0><<<dim3(8, 160), blk, 0, stream>>>(featb, wfc1, nullptr, s1b,
        fc1b, nullptr, nullptr, M, 512, 2048, 0, 0, 0, 512, 0, 0, 1);
    // fc2 (bias+relu) -> s2f
    gemm_wmma_kernel<0><<<dim3(2, 160), blk, 0, stream>>>(s1b, wfc2, s2f, nullptr,
        fc2b, nullptr, nullptr, M, 128, 512, 0, 0, 0, 128, 0, 0, 1);

    // ---- tail ----
    scores_kernel<<<64, 32, 0, stream>>>(s2f, fc3w, fc3b, scor, out + 4014144);
    mu_kernel<<<dim3(8, 640), 256, 0, stream>>>(featf, mu);
    af2_kernel<<<dim3(8, 1024), 256, 0, stream>>>(featf, af2);
    diff_kernel<<<dim3(32), dim3(32, 32), 0, stream>>>(af2, scor, idxb, out + 0, out + 32);
    topk4_kernel<<<dim3(320, 2), 256, 0, stream>>>(featf, mu, out);
    gather_kernel<<<dim3(960, 2), 256, 0, stream>>>(featf, mu, idxb, out);
}
